// HopfieldLayer_47261820125411
// MI455X (gfx1250) — matrix operand order repaired and checked by reading, not compiled, not run on hardware
//
#include <hip/hip_runtime.h>
#include <stddef.h>

// ---------------- types for WMMA ----------------
typedef __attribute__((ext_vector_type(16))) __bf16 v16bf;
typedef __attribute__((ext_vector_type(8)))  float  v8f;

union Frag16 {
    v16bf v;
    unsigned u[8];
    unsigned short h[16];
};

#define BATCH 128
#define DDIM  512
#define HDIM  1024
#define NX    (BATCH * DDIM)   // 65536
#define NH    (BATCH * HDIM)   // 131072
#define DT_C    5.0f
#define TOL_C   0.1f
#define EPS_RMS 1e-3f
#define DECAY1M 0.1f           // (1 - 0.9)

__device__ __forceinline__ unsigned short f32_to_bf16(float f) {
    unsigned u = __float_as_uint(f);
    unsigned r = u + 0x7FFFu + ((u >> 16) & 1u);   // round-to-nearest-even
    return (unsigned short)(r >> 16);
}

// ---------------- weight packing ----------------
// Pack a logical B-matrix [K x N] (bf16) into fragment-ordered tiles for
// v_wmma_f32_16x16x32_bf16. Tile (nt,kt) is 512 contiguous bf16; lane l owns
// h[e] with n = nt*16 + (l&15), k = kt*32 + (l>>4)*16 + e.
// transpose==0: Bmat[k][n] = W[k*N + n]   (W is K x N row-major)
// transpose==1: Bmat[k][n] = W[n*K + k]   (W is N x K row-major; we use W^T)
__global__ __launch_bounds__(256) void pack_weight_k(const float* __restrict__ W,
                                                     unsigned short* __restrict__ out,
                                                     int K, int N, int transpose) {
    int idx = blockIdx.x * 256 + threadIdx.x;
    if (idx >= K * N) return;
    int tile = idx >> 9;
    int r    = idx & 511;
    int l    = r >> 4;
    int e    = r & 15;
    int ktn  = K >> 5;
    int kt   = tile % ktn;
    int nt   = tile / ktn;
    int k    = kt * 32 + (l >> 4) * 16 + e;
    int n    = nt * 16 + (l & 15);
    float w  = transpose ? W[(size_t)n * K + k] : W[(size_t)k * N + n];
    out[idx] = f32_to_bf16(w);
}

// ---------------- WMMA GEMM with fused epilogues ----------------
// C[M=128, N] = A[128, K] (bf16 row-major) @ Bpack (fragment-ordered)
// EPI 0: t = tanh(acc + bias[n]);  outb = bf16(t); outf = 1 - t*t
// EPI 1: outf = acc + bias[n]                       (y)
// EPI 2: outb = bf16(acc * aux[idx])                (g_h = g_a * (1-a^2))
// EPI 3: v = -acc; outf = v; outf2 = v/(sqrt(aux)+eps); outf3 = 0.1*(v*v-aux)
template <int EPI>
__global__ __launch_bounds__(256) void gemm_wmma_k(
    const unsigned short* __restrict__ A,
    const unsigned short* __restrict__ Bp,
    int N, int K,
    const float* __restrict__ bias,
    const float* __restrict__ aux,
    unsigned short* __restrict__ outb,
    float* __restrict__ outf,
    float* __restrict__ outf2,
    float* __restrict__ outf3,
    const int* __restrict__ done) {
    if (*done) return;                 // uniform: whole wave exits, EXEC stays all-1s
    int lane = threadIdx.x & 31;
    int wave = threadIdx.x >> 5;
    int tile = blockIdx.x * 8 + wave;  // 8 waves/block share the same nt
    int mt   = tile & 7;               // M/16 == 8
    int nt   = tile >> 3;
    int m0   = mt * 16;
    int half = lane >> 4;
    int row  = lane & 15;
    int ktn  = K >> 5;

    v8f acc = {};
    const unsigned short* Arow  = A  + (size_t)(m0 + row) * K;
    const unsigned short* Btile = Bp + (size_t)nt * ktn * 512 + lane * 16;

    for (int kt = 0; kt < ktn; ++kt) {
        Frag16 fa, fb;
        const unsigned short* ap = Arow + kt * 32;
        // A 16x32 layout (matches packed B): lanes 0-15 hold K {0..15}, lanes 16-31 {16..31}
        #pragma unroll
        for (int vv = 0; vv < 8; ++vv) {
            int kb = half * 16 + 2 * vv;
            fa.u[vv] = *(const unsigned*)(ap + kb);   // two consecutive bf16
        }
        fb.v = *(const v16bf*)(Btile + (size_t)kt * 512);  // 32B contiguous, fragment-ordered
        acc = __builtin_amdgcn_wmma_f32_16x16x32_bf16(
            /*neg_a=*/false, fa.v, /*neg_b=*/false, fb.v,
            /*c_mod=*/(short)0, acc, /*reuse_a=*/false, /*reuse_b=*/false);
    }

    int n = nt * 16 + row;  // C/D: N = lane&15
    #pragma unroll
    for (int vv = 0; vv < 8; ++vv) {
        int m = m0 + vv + half * 8;           // C/D: M = vgpr + 8*half
        size_t idx = (size_t)m * N + n;
        float a = acc[vv];
        if (EPI == 0) {
            float t = tanhf(a + bias[n]);
            outb[idx] = f32_to_bf16(t);
            outf[idx] = 1.0f - t * t;
        } else if (EPI == 1) {
            outf[idx] = a + bias[n];
        } else if (EPI == 2) {
            outb[idx] = f32_to_bf16(a * aux[idx]);
        } else {
            float vval = -a;
            float m_   = aux[idx];            // mst at eval point
            outf[idx]  = vval;                // raw velocity (for stop check)
            outf2[idx] = vval / (sqrtf(m_) + EPS_RMS);   // kx
            outf3[idx] = DECAY1M * (vval * vval - m_);   // km
        }
    }
}

// ---------------- elementwise / reduction kernels ----------------
__global__ void init_k(const float* __restrict__ x0, float* __restrict__ x,
                       float* __restrict__ ms, int* __restrict__ done,
                       unsigned* __restrict__ maxbits, int n) {
    int i = blockIdx.x * blockDim.x + threadIdx.x;
    if (i == 0) { *done = 0; *maxbits = 0u; }
    if (i < n) { x[i] = x0[i]; ms[i] = 1.0f; }
}

// xt = x + c*kx (bf16 out), mst = ms + c*km
__global__ void eval_point_k(const float* __restrict__ x, const float* __restrict__ ms,
                             const float* __restrict__ kx, const float* __restrict__ km,
                             float c, float* __restrict__ mst,
                             unsigned short* __restrict__ xtb,
                             const int* __restrict__ done, int n) {
    if (*done) return;
    int i = blockIdx.x * blockDim.x + threadIdx.x;
    if (i >= n) return;
    float xv = (c == 0.0f) ? x[i]  : x[i]  + c * kx[i];
    float mv = (c == 0.0f) ? ms[i] : ms[i] + c * km[i];
    mst[i] = mv;
    xtb[i] = f32_to_bf16(xv);
}

// per-row: s = sqrt(sum y^2); g_y = 5*y/(s+1e-8) -> bf16
__global__ __launch_bounds__(256) void rownorm_k(const float* __restrict__ y,
                                                 unsigned short* __restrict__ gyb,
                                                 const int* __restrict__ done) {
    if (*done) return;
    __shared__ float red[256];
    int r = blockIdx.x, t = threadIdx.x;
    const float* yr = y + (size_t)r * DDIM;
    float a0 = yr[t], a1 = yr[t + 256];
    red[t] = a0 * a0 + a1 * a1;
    __syncthreads();
    for (int s = 128; s > 0; s >>= 1) {
        if (t < s) red[t] += red[t + s];
        __syncthreads();
    }
    float scale = 5.0f / (sqrtf(red[0]) + 1e-8f);
    unsigned short* gr = gyb + (size_t)r * DDIM;
    gr[t]       = f32_to_bf16(a0 * scale);
    gr[t + 256] = f32_to_bf16(a1 * scale);
}

__global__ __launch_bounds__(256) void vmax_k(const float* __restrict__ v,
                                              unsigned* __restrict__ maxbits,
                                              const int* __restrict__ done, int n) {
    if (*done) return;
    __shared__ float red[256];
    int t = threadIdx.x;
    float m = 0.0f;
    for (int i = blockIdx.x * 256 + t; i < n; i += gridDim.x * 256)
        m = fmaxf(m, fabsf(v[i]));
    red[t] = m;
    __syncthreads();
    for (int s = 128; s > 0; s >>= 1) {
        if (t < s) red[t] = fmaxf(red[t], red[t + s]);
        __syncthreads();
    }
    if (t == 0) atomicMax(maxbits, __float_as_uint(red[0]));  // |v|>=0: bits monotonic
}

__global__ void done_set_k(unsigned* __restrict__ maxbits, int* __restrict__ done) {
    if (__uint_as_float(*maxbits) < TOL_C) *done = 1;
    *maxbits = 0u;
}

__global__ void update_k(float* __restrict__ x, float* __restrict__ ms,
                         const float* __restrict__ k1x, const float* __restrict__ k2x,
                         const float* __restrict__ k3x, const float* __restrict__ k4x,
                         const float* __restrict__ k1m, const float* __restrict__ k2m,
                         const float* __restrict__ k3m, const float* __restrict__ k4m,
                         const int* __restrict__ done, int n) {
    if (*done) return;
    int i = blockIdx.x * blockDim.x + threadIdx.x;
    if (i >= n) return;
    const float c = DT_C / 6.0f;
    x[i]  += c * (k1x[i] + 2.0f * k2x[i] + 2.0f * k3x[i] + k4x[i]);
    ms[i] += c * (k1m[i] + 2.0f * k2m[i] + 2.0f * k3m[i] + k4m[i]);
}

__global__ void copyout_k(const float* __restrict__ x, float* __restrict__ out, int n) {
    int i = blockIdx.x * blockDim.x + threadIdx.x;
    if (i < n) out[i] = x[i];
}

// ---------------- host driver ----------------
extern "C" void kernel_launch(void* const* d_in, const int* in_sizes, int n_in,
                              void* d_out, int out_size, void* d_ws, size_t ws_size,
                              hipStream_t stream) {
    (void)in_sizes; (void)n_in; (void)out_size; (void)ws_size;
    const float* x0 = (const float*)d_in[0];   // [128,512]
    const float* W1 = (const float*)d_in[1];   // [512,1024]
    const float* b1 = (const float*)d_in[2];   // [1024]
    const float* W2 = (const float*)d_in[3];   // [1024,512]
    const float* b2 = (const float*)d_in[4];   // [512]

    char* base = (char*)d_ws;
    size_t off = 0;
    auto alloc = [&](size_t bytes) { void* p = base + off; off = (off + bytes + 255) & ~(size_t)255; return p; };

    unsigned short* W1p  = (unsigned short*)alloc(DDIM * HDIM * 2);  // B for GEMM1
    unsigned short* W2p  = (unsigned short*)alloc(HDIM * DDIM * 2);  // B for GEMM2
    unsigned short* W2Tp = (unsigned short*)alloc(DDIM * HDIM * 2);  // B for GEMM3
    unsigned short* W1Tp = (unsigned short*)alloc(HDIM * DDIM * 2);  // B for GEMM4
    float*          x    = (float*)alloc(NX * 4);
    float*          ms   = (float*)alloc(NX * 4);
    float*          mst  = (float*)alloc(NX * 4);
    unsigned short* xtb  = (unsigned short*)alloc(NX * 2);
    unsigned short* abf  = (unsigned short*)alloc(NH * 2);
    float*          tanp = (float*)alloc(NH * 4);
    float*          y    = (float*)alloc(NX * 4);
    unsigned short* gyb  = (unsigned short*)alloc(NX * 2);
    unsigned short* ghb  = (unsigned short*)alloc(NH * 2);
    float*          vv   = (float*)alloc(NX * 4);
    float*          kx1  = (float*)alloc(NX * 4);
    float*          kx2  = (float*)alloc(NX * 4);
    float*          kx3  = (float*)alloc(NX * 4);
    float*          kx4  = (float*)alloc(NX * 4);
    float*          km1  = (float*)alloc(NX * 4);
    float*          km2  = (float*)alloc(NX * 4);
    float*          km3  = (float*)alloc(NX * 4);
    float*          km4  = (float*)alloc(NX * 4);
    int*            done = (int*)alloc(256);
    unsigned*       mxb  = (unsigned*)alloc(256);

    // One-time (per launch) weight packing: all four orientations, bf16, fragment order.
    pack_weight_k<<<(DDIM * HDIM + 255) / 256, 256, 0, stream>>>(W1, W1p,  DDIM, HDIM, 0);
    pack_weight_k<<<(HDIM * DDIM + 255) / 256, 256, 0, stream>>>(W2, W2p,  HDIM, DDIM, 0);
    pack_weight_k<<<(DDIM * HDIM + 255) / 256, 256, 0, stream>>>(W2, W2Tp, DDIM, HDIM, 1);
    pack_weight_k<<<(HDIM * DDIM + 255) / 256, 256, 0, stream>>>(W1, W1Tp, HDIM, DDIM, 1);
    init_k<<<(NX + 255) / 256, 256, 0, stream>>>(x0, x, ms, done, mxb, NX);

    // tiles = 8 * (N/16); 8 waves (=1 block of 256) per 8 tiles
    const int GRID_H = (8 * (HDIM / 16)) / 8;  // 64 blocks, N=1024
    const int GRID_D = (8 * (DDIM / 16)) / 8;  // 32 blocks, N=512

    auto pvf = [&](float* kx, float* km) {
        // h=xt@W1+b1; a=tanh -> abf, tanp=1-a^2
        gemm_wmma_k<0><<<GRID_H, 256, 0, stream>>>(xtb, W1p, HDIM, DDIM, b1, nullptr,
                                                   abf, tanp, nullptr, nullptr, done);
        // y = a@W2 + b2
        gemm_wmma_k<1><<<GRID_D, 256, 0, stream>>>(abf, W2p, DDIM, HDIM, b2, nullptr,
                                                   nullptr, y, nullptr, nullptr, done);
        // g_y = 5*y/(||y||+1e-8)
        rownorm_k<<<BATCH, 256, 0, stream>>>(y, gyb, done);
        // g_h = (g_y@W2^T) * (1-a^2)
        gemm_wmma_k<2><<<GRID_H, 256, 0, stream>>>(gyb, W2Tp, HDIM, DDIM, nullptr, tanp,
                                                   ghb, nullptr, nullptr, nullptr, done);
        // v = -(g_h@W1^T); kx = v/(sqrt(mst)+1e-3); km = 0.1*(v*v - mst)
        gemm_wmma_k<3><<<GRID_D, 256, 0, stream>>>(ghb, W1Tp, DDIM, HDIM, nullptr, mst,
                                                   nullptr, vv, kx, km, done);
    };

    const int EB = (NX + 255) / 256;
    for (int step = 0; step < 20; ++step) {
        // k1 at (x, ms); raw v also drives the global stop latch (pvf(x) < tol)
        eval_point_k<<<EB, 256, 0, stream>>>(x, ms, kx1, km1, 0.0f, mst, xtb, done, NX);
        pvf(kx1, km1);
        vmax_k<<<64, 256, 0, stream>>>(vv, mxb, done, NX);
        done_set_k<<<1, 1, 0, stream>>>(mxb, done);   // latch before applying the step

        eval_point_k<<<EB, 256, 0, stream>>>(x, ms, kx1, km1, 0.5f * DT_C, mst, xtb, done, NX);
        pvf(kx2, km2);
        eval_point_k<<<EB, 256, 0, stream>>>(x, ms, kx2, km2, 0.5f * DT_C, mst, xtb, done, NX);
        pvf(kx3, km3);
        eval_point_k<<<EB, 256, 0, stream>>>(x, ms, kx3, km3, DT_C, mst, xtb, done, NX);
        pvf(kx4, km4);

        update_k<<<EB, 256, 0, stream>>>(x, ms, kx1, kx2, kx3, kx4,
                                         km1, km2, km3, km4, done, NX);
    }

    copyout_k<<<EB, 256, 0, stream>>>(x, (float*)d_out, NX);
}
